// ImageQueryGraphConvolution_15144054686141
// MI455X (gfx1250) — matrix operand order repaired and checked by reading, not compiled, not run on hardware
//
#include <hip/hip_runtime.h>

// ---------------------------------------------------------------------------
// ImageQueryGraphConvolution fused kernel for MI455X (gfx1250, wave32).
//
// Stage 0: gaussian mixture weights w[bq][n][k]          (VALU, exp)
// Stage 1: agg[bq][k][d] = sum_n w*feats   (fp32 VALU, streamed from HBM once,
//          non-temporal: 680 MB single-use stream, keep L2 for conv_w)
// Stage 2: out[bq][k*128+o] = agg @ conv_w (bf16 WMMA 16x16x32, B fragments
//          preloaded into an 8-entry register array so all 16 global b128
//          loads clause together; stage-1 FMAs hide the L2 latency)
//
// Memory-bound at ~680MB / 23.3 TB/s ~= 29us; stage-2 9.7 GFLOP rides the
// bf16 WMMA path underneath the stream.
// ---------------------------------------------------------------------------

#define BQ_TOTAL 4608
#define NN       36
#define DIM      1024
#define NK       8
#define OUTK     128
#define OUT_DIM  1024
#define TM       16            // bq rows per workgroup (== WMMA M)
#define KT_STEPS 32            // 1024 / 32
#define AGG_PAD  40            // LDS row stride (elems) for a 32-wide K chunk

typedef __attribute__((ext_vector_type(16))) __bf16 v16bf;
typedef __attribute__((ext_vector_type(8)))  float  v8f;
typedef __attribute__((ext_vector_type(2)))  float  v2f;

union Frag16 { v16bf v; uint4 u[2]; };

__device__ __forceinline__ unsigned short bf16b(float f) {
  __bf16 h = (__bf16)f;
  return __builtin_bit_cast(unsigned short, h);
}

// Pack conv_w (8,1024,128) f32 -> bf16 fragments in the per-lane WMMA B
// layout: frag(k,kt,nt)[lane][j] = conv_w[k][kt*32 + (lane>=16)*16 + j][nt*16 + (lane&15)]
__global__ __launch_bounds__(256)
void iqgc_pack_w(const float* __restrict__ conv_w, __bf16* __restrict__ wsb) {
  int idx  = blockIdx.x * 256 + threadIdx.x;   // 0 .. 1048575
  int j    =  idx        & 15;
  int lane = (idx >> 4)  & 31;
  int nt   = (idx >> 9)  & 7;
  int kt   = (idx >> 12) & 31;
  int k    = (idx >> 17);
  int d = kt * 32 + (lane >> 4) * 16 + j;      // K row of B
  int o = nt * 16 + (lane & 15);               // N column
  wsb[idx] = (__bf16)conv_w[(k * DIM + d) * OUTK + o];
}

__global__ __launch_bounds__(256)
void iqgc_fused(const float* __restrict__ feats,
                const float* __restrict__ coords,
                const float* __restrict__ mrho,
                const float* __restrict__ mtheta,
                const float* __restrict__ prho,
                const float* __restrict__ ptheta,
                const __bf16* __restrict__ wsb,
                float* __restrict__ out) {
  __shared__ float  w_lds[TM * NN * NK];                 // 18.0 KB
  __shared__ __bf16 aggbuf[2][NK * TM * AGG_PAD];        // 2 x 10.0 KB

  const int tid     = threadIdx.x;
  const int bq_base = blockIdx.x * TM;
  const int lane    = tid & 31;
  const int wv      = tid >> 5;          // wave id == gaussian-kernel id

  // ---------------- stage 0: gaussian mixture weights --------------------
  float mr[NK], mt[NK], vr[NK], vt[NK];
#pragma unroll
  for (int k = 0; k < NK; ++k) {
    mr[k] = mrho[k];
    mt[k] = mtheta[k];
    float a = prho[k];   vr[k] = -0.5f / (1e-14f + a * a);
    float b = ptheta[k]; vt[k] = -0.5f / (1e-14f + b * b);
  }
  const float TWO_PI = 6.283185307179586f;
  for (int p = tid; p < TM * NN; p += 256) {
    int m  = p / NN;
    int n  = p % NN;
    int bq = bq_base + m;
    float rho = coords[(size_t)(bq * NN + n) * 2 + 0];
    float th  = coords[(size_t)(bq * NN + n) * 2 + 1];
    float wloc[NK];
    float s = 0.f;
#pragma unroll
    for (int k = 0; k < NK; ++k) {
      float dr  = rho - mr[k];
      float fa  = fabsf(th - mt[k]);
      float sa  = fabsf(TWO_PI - fa);
      float ang = fminf(fa, sa);
      float w   = __expf(dr * dr * vr[k] + ang * ang * vt[k]);
      w = (w != w) ? 0.f : w;          // NaN -> 0 like reference
      wloc[k] = w;
      s += w;
    }
    float inv = 1.f / s;               // normalize across kernels
#pragma unroll
    for (int k = 0; k < NK; ++k) w_lds[p * NK + k] = wloc[k] * inv;
  }
  __syncthreads();

  // ---------------- stage 1 worker: one 32-wide K chunk of agg -----------
  const int sm = tid >> 4;             // bq row 0..15
  const int dp = tid & 15;             // d-pair within chunk
  const float* fbase =
      feats + ((size_t)(bq_base + sm) * NN) * DIM + dp * 2;

  auto compute_chunk = [&](int kt, int buf) {
    float2 acc[NK];
#pragma unroll
    for (int k = 0; k < NK; ++k) { acc[k].x = 0.f; acc[k].y = 0.f; }
    const float* fp = fbase + kt * 32;
#pragma unroll 4
    for (int n = 0; n < NN; ++n) {
      // single-use 680 MB stream: non-temporal, bypass L2 retention
      v2f f = __builtin_nontemporal_load(
          (const v2f*)(fp + (size_t)n * DIM));
      float4 wa = *(const float4*)&w_lds[(sm * NN + n) * NK + 0];
      float4 wb = *(const float4*)&w_lds[(sm * NN + n) * NK + 4];
      acc[0].x += f.x * wa.x; acc[0].y += f.y * wa.x;
      acc[1].x += f.x * wa.y; acc[1].y += f.y * wa.y;
      acc[2].x += f.x * wa.z; acc[2].y += f.y * wa.z;
      acc[3].x += f.x * wa.w; acc[3].y += f.y * wa.w;
      acc[4].x += f.x * wb.x; acc[4].y += f.y * wb.x;
      acc[5].x += f.x * wb.y; acc[5].y += f.y * wb.y;
      acc[6].x += f.x * wb.z; acc[6].y += f.y * wb.z;
      acc[7].x += f.x * wb.w; acc[7].y += f.y * wb.w;
    }
#pragma unroll
    for (int k = 0; k < NK; ++k) {
      ushort2 pk2;
      pk2.x = bf16b(acc[k].x);
      pk2.y = bf16b(acc[k].y);
      *(ushort2*)&aggbuf[buf][(k * TM + sm) * AGG_PAD + dp * 2] = pk2;
    }
  };

  // ---------------- stage 2: per-wave GEMM via bf16 WMMA ------------------
  compute_chunk(0, 0);
  __syncthreads();

  v8f acc[8] = {};                              // 8 N-tiles of 16 columns

  // A-fragment address: row M = lane&15, K-half by lane>=16 (contiguous K order, same as the B pack)
  const int arow = (wv * TM + (lane & 15)) * AGG_PAD + (lane >> 4) * 16;

  for (int kt = 0; kt < KT_STEPS; ++kt) {
    const int cur = kt & 1;

    // A fragment for this K chunk (conflict-free: 40-elem padded rows)
    Frag16 A;
    A.u[0] = *(const uint4*)&aggbuf[cur][arow];
    A.u[1] = *(const uint4*)&aggbuf[cur][arow + 8];

    // All 8 B fragments into distinct registers -> one load clause,
    // single wait, back-to-back WMMAs.
    const __bf16* bp =
        wsb + ((size_t)((wv * KT_STEPS + kt) * 8) * 32 + lane) * 16;
    Frag16 Bf[8];
#pragma unroll
    for (int nt = 0; nt < 8; ++nt) {
      Bf[nt].u[0] = *(const uint4*)(bp + (size_t)nt * 512);
      Bf[nt].u[1] = *(const uint4*)(bp + (size_t)nt * 512 + 8);
    }

    // Stage-1 FMA work for the next chunk hides the B-fragment L2 latency.
    if (kt + 1 < KT_STEPS) compute_chunk(kt + 1, 1 - cur);

#pragma unroll
    for (int nt = 0; nt < 8; ++nt) {
      acc[nt] = __builtin_amdgcn_wmma_f32_16x16x32_bf16(
          false, A.v, false, Bf[nt].v, (short)0, acc[nt], false, false);
    }

    __syncthreads();
  }

  // ---------------- epilogue: C layout (VGPR i -> row i / i+8) ------------
#pragma unroll
  for (int nt = 0; nt < 8; ++nt) {
#pragma unroll
    for (int i = 0; i < 8; ++i) {
      int m = i + (lane >> 4) * 8;
      __builtin_nontemporal_store(
          acc[nt][i],
          &out[(size_t)(bq_base + m) * OUT_DIM + wv * OUTK + nt * 16 +
               (lane & 15)]);
    }
  }
}

extern "C" void kernel_launch(void* const* d_in, const int* in_sizes, int n_in,
                              void* d_out, int out_size, void* d_ws, size_t ws_size,
                              hipStream_t stream) {
  (void)in_sizes; (void)n_in; (void)out_size; (void)ws_size;
  const float* feats  = (const float*)d_in[0];
  const float* coords = (const float*)d_in[1];
  const float* mrho   = (const float*)d_in[2];
  const float* mtheta = (const float*)d_in[3];
  const float* prho   = (const float*)d_in[4];
  const float* ptheta = (const float*)d_in[5];
  const float* conv_w = (const float*)d_in[6];
  __bf16* wsb = (__bf16*)d_ws;                 // 2 MB bf16 packed conv_w
  float*  out = (float*)d_out;

  iqgc_pack_w<<<4096, 256, 0, stream>>>(conv_w, wsb);
  iqgc_fused<<<BQ_TOTAL / TM, 256, 0, stream>>>(
      feats, coords, mrho, mtheta, prho, ptheta, wsb, out);
}
